// ShiftedWindowAttention_31078383354572
// MI455X (gfx1250) — matrix operand order repaired and checked by reading, not compiled, not run on hardware
//
#include <hip/hip_runtime.h>
#include <hip/hip_bf16.h>

typedef __attribute__((ext_vector_type(16))) __bf16 v16bf;
typedef __attribute__((ext_vector_type(8)))  float  v8f;

// float -> bf16 bits, round-to-nearest-even (range-safe for attention logits)
__device__ __forceinline__ unsigned short f2bf(float f) {
    unsigned int u = __float_as_uint(f);
    u = (u + 0x7FFFu + ((u >> 16) & 1u)) >> 16;
    return (unsigned short)u;
}

union FragU { v16bf v; uint4 q[2]; };

__device__ __forceinline__ v16bf ld_frag_a(const unsigned short* row, int g) {
    // A-matrix 16x32 bf16 per-lane layout: elems 0-7 = K[8g..8g+7], elems 8-15 = K[16+8g..23+8g]
    FragU u;
    u.q[0] = *(const uint4*)(row + 8 * g);
    u.q[1] = *(const uint4*)(row + 16 + 8 * g);
    return u.v;
}

__device__ __forceinline__ v16bf ld_frag_b(const unsigned short* row, int g) {
    // B-matrix 32x16 bf16 per-lane layout: elems 0-7 = K[8g..8g+7], elems 8-15 = K[16+8g..23+8g] of column (lane&15)
    FragU u;
    u.q[0] = *(const uint4*)(row + 8 * g);
    u.q[1] = *(const uint4*)(row + 16 + 8 * g);
    return u.v;
}

__device__ __forceinline__ int region_id(int h, int w) {
    // H=W=64, ws=8, ss=4: slices [0,56),[56,60),[60,64)
    int rh = (h < 56) ? 0 : ((h < 60) ? 1 : 2);
    int rw = (w < 56) ? 0 : ((w < 60) ? 1 : 2);
    return rh * 3 + rw;
}

// One block = one (window, head). 128 threads = 4 wave32s; wave w owns rows 16w..16w+15.
__global__ __launch_bounds__(128)
void swin_attn_kernel(const float* __restrict__ q,
                      const float* __restrict__ k,
                      const float* __restrict__ v,
                      const float* __restrict__ bias_table,
                      float* __restrict__ out) {
    constexpr int WS    = 8;    // window size
    constexpr int SS    = 4;    // shift
    constexpr int NTOK  = 64;   // ws*ws
    constexpr int HD    = 32;   // head dim
    constexpr int HEADS = 16;
    constexpr int HW    = 64;   // H = W
    constexpr int C     = 512;
    constexpr float SCALE = 0.17677669529663687f; // 32^-0.5

    // LDS staging (bf16 stored as ushort; strides keep 16B alignment for b128 frag loads)
    __shared__ alignas(16) unsigned short Qs[NTOK * 40];   // 64 rows x 32 dims, stride 40
    __shared__ alignas(16) unsigned short Ks[NTOK * 40];   // 64 rows x 32 dims
    __shared__ alignas(16) unsigned short Vts[HD * 72];    // V^T: 32 dims x 64 tokens, stride 72
    __shared__ alignas(16) unsigned short Ps[NTOK * 72];   // softmax probs bf16, stride 72
    __shared__ float Ss[NTOK * 68];                        // f32 logits, stride 68
    __shared__ float Bt[225];                              // rel-pos bias for this head (15*15)

    const int t    = threadIdx.x;
    const int lane = t & 31;
    const int wv   = t >> 5;       // wave id 0..3
    const int m    = lane & 15;    // sub-lane (M for A, N for B/C)
    const int g    = lane >> 4;    // lane group

    const int bid  = blockIdx.x;
    const int head = bid & (HEADS - 1);
    const int win  = bid >> 4;
    const int b    = win >> 6;
    const int wy   = (win >> 3) & 7;
    const int wx   = win & 7;

    // ---- stage bias table slice for this head ----
    for (int i = t; i < 225; i += 128) Bt[i] = bias_table[i * HEADS + head];

    // ---- load Q,K,V window (with cyclic shift folded into addressing), fp32->bf16 ----
    {
        const int n    = t >> 1;        // token 0..63 (2 threads per token)
        const int half = t & 1;         // dims [0,16) or [16,32)
        const int i    = n >> 3, j = n & 7;
        const int gh   = (wy * WS + i + SS) & (HW - 1);
        const int gw   = (wx * WS + j + SS) & (HW - 1);
        const long base = ((long)(b * HW + gh) * HW + gw) * C + head * HD + half * 16;

        const float4* qp = (const float4*)(q + base);
        const float4* kp = (const float4*)(k + base);
        const float4* vp = (const float4*)(v + base);
        float qv[16], kv[16], vv[16];
        #pragma unroll
        for (int c4 = 0; c4 < 4; ++c4) {
            float4 a = qp[c4]; qv[c4*4+0]=a.x; qv[c4*4+1]=a.y; qv[c4*4+2]=a.z; qv[c4*4+3]=a.w;
            float4 bb = kp[c4]; kv[c4*4+0]=bb.x; kv[c4*4+1]=bb.y; kv[c4*4+2]=bb.z; kv[c4*4+3]=bb.w;
            float4 cc = vp[c4]; vv[c4*4+0]=cc.x; vv[c4*4+1]=cc.y; vv[c4*4+2]=cc.z; vv[c4*4+3]=cc.w;
        }
        unsigned short* qrow = Qs + n * 40 + half * 16;
        unsigned short* krow = Ks + n * 40 + half * 16;
        #pragma unroll
        for (int e = 0; e < 16; ++e) {
            qrow[e] = f2bf(qv[e] * SCALE);
            krow[e] = f2bf(kv[e]);
            Vts[(half * 16 + e) * 72 + n] = f2bf(vv[e]);   // transposed store
        }
    }
    __syncthreads();

    // ---- S strip = Q_strip (16x32) x K^T (32x64): 4 WMMA tiles per wave ----
    {
        v16bf afrag = ld_frag_a(Qs + (16 * wv + m) * 40, g);
        #pragma unroll
        for (int t4 = 0; t4 < 4; ++t4) {
            v16bf bfrag = ld_frag_b(Ks + (16 * t4 + m) * 40, g);
            v8f acc = {};
            acc = __builtin_amdgcn_wmma_f32_16x16x32_bf16(
                false, afrag, false, bfrag, (short)0, acc, false, false);
            const int col = 16 * t4 + m;
            #pragma unroll
            for (int r = 0; r < 8; ++r)
                Ss[(16 * wv + r + 8 * g) * 68 + col] = acc[r];
        }
    }
    __syncthreads();

    // ---- bias + shift-mask + softmax, one row per thread (rows 0..63) ----
    if (t < NTOK) {
        const int n  = t;
        const int yi = n >> 3, xi = n & 7;
        const int ri = region_id(wy * WS + yi, wx * WS + xi);
        float* srow = Ss + n * 68;

        float mx = -1e30f;
        #pragma unroll 8
        for (int c2 = 0; c2 < NTOK; ++c2) {
            const int yj = c2 >> 3, xj = c2 & 7;
            float x = srow[c2] + Bt[(yi - yj + 7) * 15 + (xi - xj + 7)];
            if (region_id(wy * WS + yj, wx * WS + xj) != ri) x -= 100.0f;
            srow[c2] = x;
            mx = fmaxf(mx, x);
        }
        float sum = 0.0f;
        #pragma unroll 8
        for (int c2 = 0; c2 < NTOK; ++c2) {
            float e = __expf(srow[c2] - mx);
            srow[c2] = e;
            sum += e;
        }
        const float inv = 1.0f / sum;
        unsigned short* prow = Ps + n * 72;
        #pragma unroll 8
        for (int c2 = 0; c2 < NTOK; ++c2) prow[c2] = f2bf(srow[c2] * inv);
    }
    __syncthreads();

    // ---- O strip = P_strip (16x64) x V (64x32): 2 N-tiles x 2 chained K-chunks per wave ----
    {
        #pragma unroll
        for (int u = 0; u < 2; ++u) {
            v8f acc = {};
            #pragma unroll
            for (int kk = 0; kk < 2; ++kk) {
                v16bf afrag = ld_frag_a(Ps + (16 * wv + m) * 72 + 32 * kk, g);
                v16bf bfrag = ld_frag_b(Vts + (16 * u + m) * 72 + 32 * kk, g);
                acc = __builtin_amdgcn_wmma_f32_16x16x32_bf16(
                    false, afrag, false, bfrag, (short)0, acc, false, false);
            }
            // write out with window-reverse + reverse cyclic shift folded into addressing
            #pragma unroll
            for (int r = 0; r < 8; ++r) {
                const int n  = 16 * wv + r + 8 * g;       // token in window
                const int i  = n >> 3, j = n & 7;
                const int gh = (wy * WS + i + SS) & (HW - 1);
                const int gw = (wx * WS + j + SS) & (HW - 1);
                const long off = ((long)(b * HW + gh) * HW + gw) * C + head * HD + 16 * u + m;
                out[off] = acc[r];
            }
        }
    }
}

extern "C" void kernel_launch(void* const* d_in, const int* in_sizes, int n_in,
                              void* d_out, int out_size, void* d_ws, size_t ws_size,
                              hipStream_t stream) {
    (void)in_sizes; (void)n_in; (void)out_size; (void)d_ws; (void)ws_size;
    const float* q  = (const float*)d_in[0];
    const float* k  = (const float*)d_in[1];
    const float* v  = (const float*)d_in[2];
    const float* bt = (const float*)d_in[3];
    float* out = (float*)d_out;

    // 16 imgs * 64 windows * 16 heads = 16384 blocks, 128 threads (4 wave32s)
    dim3 grid(16384), block(128);
    swin_attn_kernel<<<grid, block, 0, stream>>>(q, k, v, bt, out);
}
